// FlashAttention_2078764171497
// MI455X (gfx1250) — matrix operand order repaired and checked by reading, not compiled, not run on hardware
//
#include <hip/hip_runtime.h>
#include <hip/hip_bf16.h>

#define D_MODEL 1024
#define NUM_HEADS 16
#define DK 64
#define BATCH 2
#define SEQ 2048
#define NTOK (BATCH * SEQ) // 4096

typedef __attribute__((ext_vector_type(16))) __bf16 v16bf;
typedef __attribute__((ext_vector_type(8)))  float  v8f;
typedef __attribute__((ext_vector_type(4)))  unsigned int u32x4;
typedef __attribute__((ext_vector_type(8)))  int i32x8;
typedef __attribute__((ext_vector_type(4)))  int i32x4;

union FragBF  { v16bf v; uint4 q[2]; };
union Pack16  { uint4 q[2]; __bf16 h[16]; };

// Build a 16-element bf16 fragment from two 16-byte chunks.
__device__ __forceinline__ v16bf frag2(const __bf16* p0, const __bf16* p1) {
    FragBF f;
    f.q[0] = *(const uint4*)p0;
    f.q[1] = *(const uint4*)p1;
    return f.v;
}

__device__ __forceinline__ v8f wmma_bf16(v16bf a, v16bf b, v8f c) {
    // (neg_a, A, neg_b, B, c_mod, C, reuse_a, reuse_b)
    return __builtin_amdgcn_wmma_f32_16x16x32_bf16(false, a, false, b, (short)0, c,
                                                   false, false);
}

// ---- Tensor Data Mover: 2D tile load global -> LDS (ISA 08_async_tensor §8) ----
// D# group0: [1:0]=count=1, [63:32]=lds_addr, [120:64]=global_addr, [127:126]=type=2
// D# group1: [17:16]=data_size(1=2B), [79:48]=tensor_dim0, [111:80]=tensor_dim1,
//            [127:112]=tile_dim0, [143:128]=tile_dim1, [207:160]=tensor_dim0_stride
__device__ __forceinline__ void tdm_load_2d(unsigned lds_byte_addr, const void* gaddr,
                                            unsigned tensor_d0, unsigned tensor_d1,
                                            unsigned tile_d0, unsigned tile_d1,
                                            unsigned long long stride0_elems) {
    unsigned long long ga = (unsigned long long)gaddr;
    u32x4 g0;
    g0[0] = 1u;                                   // count=1, user mode
    g0[1] = lds_byte_addr;                        // LDS destination (bytes)
    g0[2] = (unsigned)ga;                         // global_addr[31:0]
    g0[3] = (unsigned)(ga >> 32) | (2u << 30);    // global_addr[56:32] | type=2
    i32x8 g1;
    g1[0] = (int)(1u << 16);                      // data_size = 2 bytes; no multicast
    g1[1] = (int)((tensor_d0 & 0xFFFFu) << 16);   // tensor_dim0[15:0]
    g1[2] = (int)((tensor_d0 >> 16) | ((tensor_d1 & 0xFFFFu) << 16));
    g1[3] = (int)((tensor_d1 >> 16) | (tile_d0 << 16));
    g1[4] = (int)tile_d1;                         // tile_dim1; tile_dim2=0
    g1[5] = (int)(unsigned)stride0_elems;         // tensor_dim0_stride[31:0]
    g1[6] = (int)(unsigned)(stride0_elems >> 32); // stride[47:32]; dim1_stride=0
    g1[7] = 0;
    i32x4 gz = {0, 0, 0, 0};
#if defined(__clang_major__) && (__clang_major__ >= 23)
    i32x8 gz8 = {0, 0, 0, 0, 0, 0, 0, 0};
    __builtin_amdgcn_tensor_load_to_lds(g0, g1, gz, gz, gz8, 0);
#else
    __builtin_amdgcn_tensor_load_to_lds(g0, g1, gz, gz, 0);
#endif
}

__device__ __forceinline__ unsigned lds_addr_of(const void* p) {
    return (unsigned)(unsigned long long)p;   // low 32 bits = LDS offset
}

// Transpose-convert: Wt[n][k] = (bf16)W[k][n]. Tiled through LDS so both the
// global read and the global write stay coalesced.
__global__ __launch_bounds__(256)
void cvt_transpose_f32_bf16(const float* __restrict__ W, __bf16* __restrict__ Wt) {
    __shared__ float tile[32][33];
    const int tx = threadIdx.x & 31;
    const int ty = threadIdx.x >> 5;          // 0..7
    const int nb = blockIdx.x * 32;
    const int kb = blockIdx.y * 32;
    for (int i = 0; i < 4; ++i)
        tile[ty + i * 8][tx] = W[(size_t)(kb + ty + i * 8) * D_MODEL + nb + tx];
    __syncthreads();
    for (int i = 0; i < 4; ++i)
        Wt[(size_t)(nb + ty + i * 8) * D_MODEL + kb + tx] = (__bf16)tile[tx][ty + i * 8];
}

// C[4096,1024] = A[4096,1024] @ Wt^T + bias  (Wt is [out=n][in=k], bf16).
// Double-buffered LDS; B (and A when bf16) staged by the Tensor Data Mover so
// tile k+1 DMA overlaps the 8 WMMAs on tile k. Epilogue scatter by MODE:
//   MODE 0: bf16 out as [B,H,S,DK]   (Q, K)
//   MODE 1: bf16 out as [B,H,DK,S]   (V transposed)
//   MODE 2: fp32 out as [4096,1024]  (final projection)
template <int MODE, bool A_F32>
__global__ __launch_bounds__(256)
void gemm_wmma(const void* __restrict__ Ap, const __bf16* __restrict__ Wt,
               const float* __restrict__ bias, void* __restrict__ outp) {
    __shared__ __align__(16) __bf16 sA[2][128][32];   // [m][k]
    __shared__ __align__(16) __bf16 sB[2][128][32];   // [n][k]

    const int t    = threadIdx.x;
    const int lane = t & 31;
    const int wave = t >> 5;
    const int half = lane >> 4;
    const int l15  = lane & 15;
    const int wm   = wave >> 1;   // 0..3 -> m offset wm*32
    const int wn   = wave & 1;    // 0..1 -> n offset wn*64
    const int mBase = blockIdx.y * 128;
    const int nBase = blockIdx.x * 128;

    v8f zero = {0.f, 0.f, 0.f, 0.f, 0.f, 0.f, 0.f, 0.f};
    v8f acc[8];
    for (int i = 0; i < 8; ++i) acc[i] = zero;

    const int ar = t >> 1, ah = t & 1;    // fp32-A staging: 128 rows x 2 halves of 16

    // Manual fp32 -> bf16 A staging (TDM cannot convert).
    auto stageA = [&](int buf, int kk) {
        const float* src = (const float*)Ap + (size_t)(mBase + ar) * D_MODEL + kk + ah * 16;
        float f[16];
        ((float4*)f)[0] = ((const float4*)src)[0];
        ((float4*)f)[1] = ((const float4*)src)[1];
        ((float4*)f)[2] = ((const float4*)src)[2];
        ((float4*)f)[3] = ((const float4*)src)[3];
        if (kk + 32 < D_MODEL) __builtin_prefetch(src + 32, 0, 0);
        Pack16 u;
        for (int i = 0; i < 16; ++i) u.h[i] = (__bf16)f[i];
        *(uint4*)&sA[buf][ar][ah * 16]     = u.q[0];
        *(uint4*)&sA[buf][ar][ah * 16 + 8] = u.q[1];
    };

    // ---- prologue: stage k-tile 0 into buffer 0 ----
    if (A_F32) stageA(0, 0);
    if (wave == 0) {
        tdm_load_2d(lds_addr_of(&sB[0][0][0]), Wt + (size_t)nBase * D_MODEL,
                    D_MODEL, D_MODEL, 32, 128, D_MODEL);
        if (!A_F32)
            tdm_load_2d(lds_addr_of(&sA[0][0][0]), (const __bf16*)Ap + (size_t)mBase * D_MODEL,
                        D_MODEL, NTOK, 32, 128, D_MODEL);
    }

    for (int kk = 0; kk < D_MODEL; kk += 32) {
        const int cur = (kk >> 5) & 1;
        if (wave == 0) __builtin_amdgcn_s_wait_tensorcnt(0);  // tile kk landed
        __syncthreads();                                      // visible to all waves
        if (kk + 32 < D_MODEL) {                              // prefetch tile kk+32
            if (wave == 0) {
                tdm_load_2d(lds_addr_of(&sB[cur ^ 1][0][0]),
                            Wt + (size_t)nBase * D_MODEL + kk + 32,
                            D_MODEL, D_MODEL, 32, 128, D_MODEL);
                if (!A_F32)
                    tdm_load_2d(lds_addr_of(&sA[cur ^ 1][0][0]),
                                (const __bf16*)Ap + (size_t)mBase * D_MODEL + kk + 32,
                                D_MODEL, NTOK, 32, 128, D_MODEL);
            }
            if (A_F32) stageA(cur ^ 1, kk + 32);
        }

        // ---- fragments + 8 WMMAs on buffer `cur` ----
        v16bf af[2], bfr[4];
        for (int mi = 0; mi < 2; ++mi) {
            const __bf16* r0 = &sA[cur][wm * 32 + mi * 16 + l15][0];
            af[mi] = frag2(r0 + 8 * half, r0 + 16 + 8 * half);
        }
        for (int ni = 0; ni < 4; ++ni) {
            const __bf16* c0 = &sB[cur][wn * 64 + ni * 16 + l15][0];
            bfr[ni] = frag2(c0 + 8 * half, c0 + 16 + 8 * half);
        }
        for (int mi = 0; mi < 2; ++mi)
            for (int ni = 0; ni < 4; ++ni)
                acc[mi * 4 + ni] = wmma_bf16(af[mi], bfr[ni], acc[mi * 4 + ni]);
        __syncthreads();   // retire reads of buf[cur] before it is re-filled
    }

    // ---- epilogue: bias + scatter ----
    for (int mi = 0; mi < 2; ++mi)
        for (int ni = 0; ni < 4; ++ni) {
            const int colb = nBase + wn * 64 + ni * 16 + l15;
            const float bcol = bias[colb];
            for (int j = 0; j < 8; ++j) {
                const int row = mBase + wm * 32 + mi * 16 + j + 8 * half;
                float v = acc[mi * 4 + ni][j] + bcol;
                if (MODE == 2) {
                    ((float*)outp)[(size_t)row * D_MODEL + colb] = v;
                } else {
                    const int b = row >> 11, s = row & (SEQ - 1);
                    const int h = colb >> 6, d = colb & 63;
                    if (MODE == 0)
                        ((__bf16*)outp)[((size_t)(b * NUM_HEADS + h) * SEQ + s) * DK + d] = (__bf16)v;
                    else
                        ((__bf16*)outp)[((size_t)(b * NUM_HEADS + h) * DK + d) * SEQ + s] = (__bf16)v;
                }
            }
        }
}

// Flash attention. Q,K: [B,H,S,DK] bf16; Vt: [B,H,DK,S] bf16.
// Block = 64 query rows (4 waves x 16). K/V^T tiles staged by the Tensor Data
// Mover (double-buffered, wave 0 issues, TENSORcnt-tracked) so tile kt+1 DMA
// overlaps the 16 WMMAs + softmax of tile kt.
__global__ __launch_bounds__(128)
void flash_attn(const __bf16* __restrict__ Q, const __bf16* __restrict__ K,
                const __bf16* __restrict__ Vt, __bf16* __restrict__ Abuf) {
    __shared__ __align__(16) __bf16 sK[2][64][64];    // key-tile rows (s major)
    __shared__ __align__(16) __bf16 sVt[2][64][64];   // V^T tile (dk major)
    __shared__ __align__(16) __bf16 sP[4][16][64];    // per-wave P relayout buffer

    const int t    = threadIdx.x;
    const int lane = t & 31;
    const int wave = t >> 5;
    const int half = lane >> 4;
    const int l15  = lane & 15;
    const int bh   = blockIdx.y;           // b*H + h
    const int qbase = blockIdx.x * 64;

    const float scale = 0.125f;            // 1/sqrt(DK)
    const int NT = SEQ / 64;

    // Q fragments (persist across all key tiles): A-operand 16x64 = 2 k-blocks.
    const int qrow = qbase + wave * 16 + l15;
    const __bf16* qp = Q + ((size_t)bh * SEQ + qrow) * DK;
    v16bf qa[2];
    for (int kb = 0; kb < 2; ++kb)
        qa[kb] = frag2(qp + kb * 32 + 8 * half, qp + kb * 32 + 16 + 8 * half);

    v8f zero = {0.f, 0.f, 0.f, 0.f, 0.f, 0.f, 0.f, 0.f};
    v8f oacc[4];
    for (int i = 0; i < 4; ++i) oacc[i] = zero;
    float mprev[8], lrun[8];
    for (int j = 0; j < 8; ++j) { mprev[j] = -3.0e38f; lrun[j] = 0.f; }

    // ---- TDM prologue: kick off tile 0 into buffer 0 ----
    if (wave == 0) {
        tdm_load_2d(lds_addr_of(&sK[0][0][0]),
                    K + (size_t)bh * SEQ * DK,
                    DK, 64, DK, 64, DK);                    // contiguous 64x64
        tdm_load_2d(lds_addr_of(&sVt[0][0][0]),
                    Vt + (size_t)bh * DK * SEQ,
                    SEQ, DK, 64, DK, SEQ);                  // 64 rows, stride SEQ
    }

    for (int kt = 0; kt < NT; ++kt) {
        const int cur = kt & 1;
        if (wave == 0) __builtin_amdgcn_s_wait_tensorcnt(0);  // tile kt landed
        __syncthreads();                                      // visible to all waves
        if (wave == 0 && kt + 1 < NT) {                       // overlap next DMA
            tdm_load_2d(lds_addr_of(&sK[cur ^ 1][0][0]),
                        K + ((size_t)bh * SEQ + (size_t)(kt + 1) * 64) * DK,
                        DK, 64, DK, 64, DK);
            tdm_load_2d(lds_addr_of(&sVt[cur ^ 1][0][0]),
                        Vt + (size_t)bh * DK * SEQ + (size_t)(kt + 1) * 64,
                        SEQ, DK, 64, DK, SEQ);
        }

        // ---- scores S = (Q K^T) * scale : 4 n-subtiles x 2 k-blocks ----
        v8f sacc[4];
        for (int nt = 0; nt < 4; ++nt) {
            sacc[nt] = zero;
            const __bf16* c0 = &sK[cur][nt * 16 + l15][0];
            for (int kb = 0; kb < 2; ++kb) {
                v16bf bfr = frag2(c0 + kb * 32 + 8 * half, c0 + kb * 32 + 16 + 8 * half);
                sacc[nt] = wmma_bf16(qa[kb], bfr, sacc[nt]);
            }
        }
        for (int nt = 0; nt < 4; ++nt)
            for (int j = 0; j < 8; ++j) sacc[nt][j] *= scale;

        // ---- online softmax (row j+8*half lives in lanes sharing `half`) ----
        float mnew[8], alpha[8], rsum[8];
        for (int j = 0; j < 8; ++j) {
            float m = fmaxf(fmaxf(sacc[0][j], sacc[1][j]), fmaxf(sacc[2][j], sacc[3][j]));
            for (int off = 8; off; off >>= 1) m = fmaxf(m, __shfl_xor(m, off, 32));
            float mn = fmaxf(mprev[j], m);
            alpha[j] = __expf(mprev[j] - mn);
            mprev[j] = mn;
            mnew[j]  = mn;
            rsum[j]  = 0.f;
        }
        for (int nt = 0; nt < 4; ++nt)
            for (int j = 0; j < 8; ++j) {
                float p = __expf(sacc[nt][j] - mnew[j]);
                rsum[j] += p;
                sP[wave][j + 8 * half][nt * 16 + l15] = (__bf16)p;  // C->A relayout via LDS
            }
        for (int j = 0; j < 8; ++j) {
            float rs = rsum[j];
            for (int off = 8; off; off >>= 1) rs += __shfl_xor(rs, off, 32);
            lrun[j] = lrun[j] * alpha[j] + rs;
        }
        for (int nt = 0; nt < 4; ++nt)
            for (int j = 0; j < 8; ++j) oacc[nt][j] *= alpha[j];

        // ---- O += P @ V : P fragments from per-wave LDS (in-order w/ stores) ----
        v16bf pa[2];
        {
            const __bf16* pr = &sP[wave][l15][0];
            for (int kb = 0; kb < 2; ++kb)
                pa[kb] = frag2(pr + kb * 32 + 8 * half, pr + kb * 32 + 16 + 8 * half);
        }
        for (int nt = 0; nt < 4; ++nt) {
            const __bf16* c0 = &sVt[cur][nt * 16 + l15][0];
            for (int kb = 0; kb < 2; ++kb) {
                v16bf bfr = frag2(c0 + kb * 32 + 8 * half, c0 + kb * 32 + 16 + 8 * half);
                oacc[nt] = wmma_bf16(pa[kb], bfr, oacc[nt]);
            }
        }
        __syncthreads();   // retire reads of buf[cur] before TDM reuses it
    }

    // ---- epilogue: normalize and write token-major bf16 ----
    const int b = bh >> 4, h = bh & 15;
    for (int j = 0; j < 8; ++j) {
        const float inv = 1.0f / lrun[j];
        const int s = qbase + wave * 16 + j + 8 * half;
        for (int nt = 0; nt < 4; ++nt) {
            const int n = nt * 16 + l15;
            Abuf[((size_t)(b * SEQ) + s) * D_MODEL + h * DK + n] = (__bf16)(oacc[nt][j] * inv);
        }
    }
}

extern "C" void kernel_launch(void* const* d_in, const int* in_sizes, int n_in,
                              void* d_out, int out_size, void* d_ws, size_t ws_size,
                              hipStream_t stream) {
    (void)in_sizes; (void)n_in; (void)out_size; (void)ws_size;
    const float* q  = (const float*)d_in[0];
    const float* k  = (const float*)d_in[1];
    const float* v  = (const float*)d_in[2];
    const float* Wq = (const float*)d_in[3];
    const float* bq = (const float*)d_in[4];
    const float* Wk = (const float*)d_in[5];
    const float* bk = (const float*)d_in[6];
    const float* Wv = (const float*)d_in[7];
    const float* bv = (const float*)d_in[8];
    const float* Wo = (const float*)d_in[9];
    const float* bo = (const float*)d_in[10];

    char* ws = (char*)d_ws;
    const size_t WSZ = (size_t)D_MODEL * D_MODEL;  // 1M elems (weights)
    const size_t TSZ = (size_t)NTOK * D_MODEL;     // 4M elems (activations)
    __bf16* Wqt = (__bf16*)(ws + 0 * WSZ);         // transposed bf16 weights [n][k]
    __bf16* Wkt = (__bf16*)(ws + 2 * WSZ);
    __bf16* Wvt = (__bf16*)(ws + 4 * WSZ);
    __bf16* Wot = (__bf16*)(ws + 6 * WSZ);
    __bf16* Qb  = (__bf16*)(ws + 8 * WSZ);
    __bf16* Kb  = (__bf16*)(ws + 8 * WSZ + 2 * TSZ);
    __bf16* Vtb = (__bf16*)(ws + 8 * WSZ + 4 * TSZ);
    __bf16* Ab  = (__bf16*)(ws + 8 * WSZ + 6 * TSZ);  // total 40 MB

    dim3 tg(D_MODEL / 32, D_MODEL / 32);  // (32, 32)
    cvt_transpose_f32_bf16<<<tg, 256, 0, stream>>>(Wq, Wqt);
    cvt_transpose_f32_bf16<<<tg, 256, 0, stream>>>(Wk, Wkt);
    cvt_transpose_f32_bf16<<<tg, 256, 0, stream>>>(Wv, Wvt);
    cvt_transpose_f32_bf16<<<tg, 256, 0, stream>>>(Wo, Wot);

    dim3 gg(D_MODEL / 128, NTOK / 128);  // (8, 32)
    gemm_wmma<0, true><<<gg, 256, 0, stream>>>(q, Wqt, bq, Qb);
    gemm_wmma<0, true><<<gg, 256, 0, stream>>>(k, Wkt, bk, Kb);
    gemm_wmma<1, true><<<gg, 256, 0, stream>>>(v, Wvt, bv, Vtb);

    flash_attn<<<dim3(SEQ / 64, BATCH * NUM_HEADS), 128, 0, stream>>>(Qb, Kb, Vtb, Ab);

    gemm_wmma<2, false><<<gg, 256, 0, stream>>>(Ab, Wot, bo, (float*)d_out);
}
